// BasicTransformerBlock_Epipolar_35888746725716
// MI455X (gfx1250) — matrix operand order repaired and checked by reading, not compiled, not run on hardware
//
#include <hip/hip_runtime.h>

typedef __attribute__((ext_vector_type(16))) __bf16 bf16x16;
typedef __attribute__((ext_vector_type(8)))  float  f32x8;

#define WMMA_BF16(a, b, c) \
  __builtin_amdgcn_wmma_f32_16x16x32_bf16(false, (a), false, (b), (short)0, (c), false, false)

__device__ __forceinline__ f32x8 zero8() {
  f32x8 z;
#pragma unroll
  for (int g = 0; g < 8; g++) z[g] = 0.0f;
  return z;
}

// fragment-major load: two contiguous 16B chunks per lane -> 2x ds_load_b128
__device__ __forceinline__ bf16x16 ldfrag(const __bf16* lo, const __bf16* hi) {
  bf16x16 f;
  *(uint4*)&f       = *(const uint4*)lo;
  *((uint4*)&f + 1) = *(const uint4*)hi;
  return f;
}

__global__ void cast_f32_to_bf16(const float* __restrict__ src, __bf16* __restrict__ dst, int n) {
  int i = blockIdx.x * blockDim.x + threadIdx.x;
  int stride = gridDim.x * blockDim.x;
  for (; i < n; i += stride) dst[i] = (__bf16)src[i];
}

// W[1024][1024] -> WT[1024][1024], WT[n][k] = W[k][n] (bf16)
__global__ void cast_transpose_1024(const float* __restrict__ W, __bf16* __restrict__ WT) {
  int i = blockIdx.x * blockDim.x + threadIdx.x;
  int stride = gridDim.x * blockDim.x;
  for (; i < (1 << 20); i += stride) {
    int n = i >> 10, k = i & 1023;
    WT[(size_t)n * 1024 + k] = (__bf16)W[(size_t)k * 1024 + n];
  }
}

// rel_k -> bf16 padded [2080][64]; rel_v -> bf16 transposed [64][2080] (zero pad r>=2049)
__global__ void prep_rel(const float* __restrict__ rk, const float* __restrict__ rv,
                         __bf16* __restrict__ rkp, __bf16* __restrict__ rvT) {
  int i = blockIdx.x * blockDim.x + threadIdx.x;
  int stride = gridDim.x * blockDim.x;
  for (; i < 2080 * 64; i += stride) {
    int r = i >> 6, d = i & 63;
    float vk = (r < 2049) ? rk[(size_t)r * 64 + d] : 0.0f;
    float vv = (r < 2049) ? rv[(size_t)r * 64 + d] : 0.0f;
    rkp[(size_t)r * 64 + d]   = (__bf16)vk;
    rvT[(size_t)d * 2080 + r] = (__bf16)vv;
  }
}

// C = alpha * A[MxK] * B, B given transposed as BT[NxK]. bf16 in, f32 accum.
// mode 0: bf16 head-split [B,H,T,dh] ; mode 1: f32 + bias ; mode 2: bf16 head-split transposed [B,H,dh,T]
__global__ __launch_bounds__(256)
void gemm_bf16_128x128(const __bf16* __restrict__ A, const __bf16* __restrict__ BT,
                       int M, int N, int K,
                       __bf16* __restrict__ out_bf, float* __restrict__ out_f,
                       const float* __restrict__ bias, float alpha, int mode) {
  __shared__ alignas(16) __bf16 Asub[8][2][32][8];  // [rowblk16][ehalf][lane][8]
  __shared__ alignas(16) __bf16 Bsub[8][2][32][8];  // [colblk16][ehalf][lane][8]

  int tid = threadIdx.x;
  int lane = tid & 31;
  int wid = tid >> 5;
  int hlf = lane >> 4, n16 = lane & 15;
  int wm = wid >> 1, wn = wid & 1;                  // 4x2 waves -> 32x64 each
  int m0 = blockIdx.y * 128, n0 = blockIdx.x * 128;

  f32x8 acc[2][4];
#pragma unroll
  for (int i = 0; i < 2; i++)
#pragma unroll
    for (int j = 0; j < 4; j++) acc[i][j] = zero8();

  for (int kk = 0; kk < K; kk += 32) {
    __syncthreads();
#pragma unroll
    for (int r = 0; r < 2; r++) {                   // A tile 128x32, frag-major
      int lin = tid + 256 * r;
      int rb16 = lin >> 6, eh = (lin >> 5) & 1, L = lin & 31;
      *(uint4*)&Asub[rb16][eh][L][0] =
          *(const uint4*)&A[(size_t)(m0 + rb16 * 16 + (L & 15)) * K + kk + eh * 16 + (L >> 4) * 8];
    }
#pragma unroll
    for (int r = 0; r < 2; r++) {                   // B tile 32x128 from BT, frag-major
      int lin = tid + 256 * r;
      int cb16 = lin >> 6, eh = (lin >> 5) & 1, L = lin & 31;
      *(uint4*)&Bsub[cb16][eh][L][0] =
          *(const uint4*)&BT[(size_t)(n0 + cb16 * 16 + (L & 15)) * K + kk + eh * 16 + (L >> 4) * 8];
    }
    if (kk + 32 < K) {
      __builtin_prefetch(&A[(size_t)(m0 + (tid >> 1)) * K + kk + 32], 0, 1);
      __builtin_prefetch(&BT[(size_t)(n0 + (tid >> 1)) * K + kk + 32], 0, 1);
    }
    __syncthreads();

    bf16x16 af[2];
#pragma unroll
    for (int i = 0; i < 2; i++)
      af[i] = ldfrag(&Asub[wm * 2 + i][0][lane][0], &Asub[wm * 2 + i][1][lane][0]);
#pragma unroll
    for (int j = 0; j < 4; j++) {
      bf16x16 bfr = ldfrag(&Bsub[wn * 4 + j][0][lane][0], &Bsub[wn * 4 + j][1][lane][0]);
#pragma unroll
      for (int i = 0; i < 2; i++) acc[i][j] = WMMA_BF16(af[i], bfr, acc[i][j]);
    }
  }

#pragma unroll
  for (int i = 0; i < 2; i++)
#pragma unroll
    for (int j = 0; j < 4; j++)
#pragma unroll
      for (int g = 0; g < 8; g++) {
        int m = m0 + wm * 32 + 16 * i + g + 8 * hlf;
        int nn = n0 + wn * 64 + 16 * j + n16;
        float v = acc[i][j][g] * alpha;
        if (mode == 0) {
          int bb = m >> 10, tt = m & 1023;
          int hh = nn >> 6, dd = nn & 63;
          out_bf[((size_t)(bb * 16 + hh) * 1024 + tt) * 64 + dd] = (__bf16)v;
        } else if (mode == 2) {
          int bb = m >> 10, tt = m & 1023;
          int hh = nn >> 6, dd = nn & 63;
          out_bf[((size_t)(bb * 16 + hh) * 64 + dd) * 1024 + tt] = (__bf16)v;
        } else {
          out_f[(size_t)m * N + nn] = v + bias[nn];
        }
      }
}

// Flash attention + relative K/V. 128 threads (4 waves), 64 q-rows, 32-key blocks.
// Q pre-scaled by dh^-0.5. V given transposed [bh][d][t]; relv transposed/padded [64][2080].
__global__ __launch_bounds__(128)
void attn_rel64x32(const __bf16* __restrict__ Qg, const __bf16* __restrict__ Kg,
                   const __bf16* __restrict__ VTg,
                   const __bf16* __restrict__ RKg, const __bf16* __restrict__ RVTg,
                   __bf16* __restrict__ Og) {
  __shared__ alignas(16) __bf16 QA[4][2][2][32][8];    // per-wave Q A-frags; [w][0] reused as probs A-frags
  __shared__ alignas(16) __bf16 KsF[2][2][2][32][8];   // B-frags for QK^T  [sblk][kc][eh][lane][8]
  __shared__ alignas(16) __bf16 VsF[4][2][32][8];      // B-frags for attn@V [dblk][eh][lane][8]
  __shared__ alignas(16) __bf16 RKsF[7][2][2][32][8];  // B-frags for Q@relk^T, 112-wide r window
  __shared__ alignas(16) __bf16 RVsF[4][4][2][32][8];  // B-frags for Pv@relv, 128-wide r window
  __shared__ alignas(16) __bf16 PvF[4][4][2][32][8];   // per-wave Pv A-frags; also P stripe [112][16] col-major

  int tid = threadIdx.x;
  int lane = tid & 31, w = tid >> 5;
  int hlf = lane >> 4, n16 = lane & 15;
  int t0 = blockIdx.x * 64;
  int bh = blockIdx.y;
  int b = bh >> 4, h = bh & 15;
  const __bf16* Qp  = Qg  + (size_t)bh * 1024 * 64;
  const __bf16* Kp  = Kg  + (size_t)bh * 1024 * 64;
  const __bf16* VTp = VTg + (size_t)bh * 64 * 1024;

  // ---- Q A-frags: frag-major load then keep in registers ----
#pragma unroll
  for (int r = 0; r < 4; r++) {
    int lin = tid + 128 * r;
    int wv = lin >> 7, kc = (lin >> 6) & 1, eh = (lin >> 5) & 1, L = lin & 31;
    *(uint4*)&QA[wv][kc][eh][L][0] =
        *(const uint4*)&Qp[(size_t)(t0 + wv * 16 + (L & 15)) * 64 + kc * 32 + (L >> 4) * 16 + eh * 8];
  }
  __syncthreads();
  bf16x16 aQ[2];
#pragma unroll
  for (int kc = 0; kc < 2; kc++)
    aQ[kc] = ldfrag(&QA[w][kc][0][lane][0], &QA[w][kc][1][lane][0]);

  __bf16* PPw = &PvF[w][0][0][0][0];   // P stripe overlay, col-major [col][16]
  auto Asw = QA[w][0];                 // probs A-frags overlay [eh][lane][8]

  float m_i[8], l_i[8];
#pragma unroll
  for (int g = 0; g < 8; g++) { m_i[g] = -1e30f; l_i[g] = 0.0f; }
  f32x8 accO[4];
#pragma unroll
  for (int jo = 0; jo < 4; jo++) accO[jo] = zero8();

  for (int s0 = 0; s0 < 1024; s0 += 32) {
    int rb = s0 - t0 + 961;            // r = s-t+1024 window base, in [1,1953]
    int rb8 = rb & ~7;                 // 16B-aligned window base
    int delta = rb - rb8;              // 0..7
    __syncthreads();
#pragma unroll
    for (int r = 0; r < 2; r++) {      // K tile 32x64 -> frag-major
      int lin = tid + 128 * r;
      int jb = (lin >> 7) & 1, kc = (lin >> 6) & 1, eh = (lin >> 5) & 1, L = lin & 31;
      *(uint4*)&KsF[jb][kc][eh][L][0] =
          *(const uint4*)&Kp[(size_t)(s0 + jb * 16 + (L & 15)) * 64 + kc * 32 + (L >> 4) * 16 + eh * 8];
    }
#pragma unroll
    for (int r = 0; r < 2; r++) {      // V^T tile -> frag-major (K-dim = s)
      int lin = tid + 128 * r;
      int jb = (lin >> 6) & 3, eh = (lin >> 5) & 1, L = lin & 31;
      *(uint4*)&VsF[jb][eh][L][0] =
          *(const uint4*)&VTp[(size_t)(jb * 16 + (L & 15)) * 1024 + s0 + eh * 16 + (L >> 4) * 8];
    }
#pragma unroll
    for (int r = 0; r < 7; r++) {      // relk window [rb8, rb8+112)
      int lin = tid + 128 * r;
      int jjb = lin >> 7, kc = (lin >> 6) & 1, eh = (lin >> 5) & 1, L = lin & 31;
      *(uint4*)&RKsF[jjb][kc][eh][L][0] =
          *(const uint4*)&RKg[(size_t)(rb8 + jjb * 16 + (L & 15)) * 64 + kc * 32 + (L >> 4) * 16 + eh * 8];
    }
#pragma unroll
    for (int r = 0; r < 8; r++) {      // relv^T window [rb8, rb8+128) (K-dim = r)
      int lin = tid + 128 * r;
      int jb = lin >> 8, kc = (lin >> 6) & 3, eh = (lin >> 5) & 1, L = lin & 31;
      *(uint4*)&RVsF[jb][kc][eh][L][0] =
          *(const uint4*)&RVTg[(size_t)(jb * 16 + (L & 15)) * 2080 + rb8 + kc * 32 + eh * 16 + (L >> 4) * 8];
    }
    __syncthreads();

    // --- sim = Q @ K^T ---
    f32x8 accS[2];
    accS[0] = zero8(); accS[1] = zero8();
#pragma unroll
    for (int jb = 0; jb < 2; jb++)
#pragma unroll
      for (int kc = 0; kc < 2; kc++) {
        bf16x16 bfr = ldfrag(&KsF[jb][kc][0][lane][0], &KsF[jb][kc][1][lane][0]);
        accS[jb] = WMMA_BF16(aQ[kc], bfr, accS[jb]);
      }

    // --- P = Q @ relk^T over 112-wide window; packed b128 spill, col-major [col][16] ---
#pragma unroll
    for (int jjb = 0; jjb < 7; jjb++) {
      f32x8 accP = zero8();
#pragma unroll
      for (int kc = 0; kc < 2; kc++) {
        bf16x16 bfr = ldfrag(&RKsF[jjb][kc][0][lane][0], &RKsF[jjb][kc][1][lane][0]);
        accP = WMMA_BF16(aQ[kc], bfr, accP);
      }
      uint4 pk;
      __bf16* ph = (__bf16*)&pk;
#pragma unroll
      for (int g = 0; g < 8; g++) ph[g] = (__bf16)accP[g];
      *(uint4*)&PPw[(16 * jjb + n16) * 16 + 8 * hlf] = pk;
    }

    // --- diagonal gather: sim[i,s] += P[s - t + 63 + delta][i] ---
#pragma unroll
    for (int j = 0; j < 2; j++)
#pragma unroll
      for (int g = 0; g < 8; g++) {
        int i = g + 8 * hlf;
        int rl = 16 * j + n16 - (16 * w + i) + 63 + delta;
        accS[j][g] += (float)PPw[rl * 16 + i];
      }

    // --- zero Pv frag buffer (b128), then scatter probs diagonally ---
    {
      uint4 z = {0u, 0u, 0u, 0u};
      __bf16* pvw = PPw;
      for (int c = lane; c < 256; c += 32) *(uint4*)&pvw[c * 8] = z;
    }

    // --- online softmax + prob stores ---
#pragma unroll
    for (int g = 0; g < 8; g++) {
      float v0 = accS[0][g], v1 = accS[1][g];
      float mx = fmaxf(v0, v1);
      mx = fmaxf(mx, __shfl_xor(mx, 1, 16));
      mx = fmaxf(mx, __shfl_xor(mx, 2, 16));
      mx = fmaxf(mx, __shfl_xor(mx, 4, 16));
      mx = fmaxf(mx, __shfl_xor(mx, 8, 16));
      float mn = fmaxf(m_i[g], mx);
      float p0 = __expf(v0 - mn), p1 = __expf(v1 - mn);
      float sc = __expf(m_i[g] - mn);
      float ls = p0 + p1;
      ls += __shfl_xor(ls, 1, 16);
      ls += __shfl_xor(ls, 2, 16);
      ls += __shfl_xor(ls, 4, 16);
      ls += __shfl_xor(ls, 8, 16);
      l_i[g] = l_i[g] * sc + ls;
      m_i[g] = mn;
#pragma unroll
      for (int jo = 0; jo < 4; jo++) accO[jo][g] *= sc;

      int i = g + 8 * hlf;
      int Lp = i + 16 * (n16 >> 3);              // probs -> A-frag slots
      Asw[0][Lp][n16 & 7] = (__bf16)p0;          // key = n16
      Asw[1][Lp][n16 & 7] = (__bf16)p1;          // key = 16 + n16
      // diagonal scatter into Pv A-frags: K index rl = s - t + 63 + delta
      int rl0 = n16 - (16 * w + i) + 63 + delta;
#pragma unroll
      for (int pp = 0; pp < 2; pp++) {
        int rl = rl0 + 16 * pp;
        int kc = rl >> 5, k32 = rl & 31;
        int eh = (k32 >> 4) & 1, Lh = (k32 >> 3) & 1, e7 = k32 & 7;
        PvF[w][kc][eh][i + (Lh << 4)][e7] = (__bf16)(pp ? p1 : p0);
      }
    }

    // --- out += attn @ V ---
    {
      bf16x16 aP = ldfrag(&Asw[0][lane][0], &Asw[1][lane][0]);
#pragma unroll
      for (int jo = 0; jo < 4; jo++) {
        bf16x16 bfr = ldfrag(&VsF[jo][0][lane][0], &VsF[jo][1][lane][0]);
        accO[jo] = WMMA_BF16(aP, bfr, accO[jo]);
      }
    }
    // --- out += Pv @ relv[window]  (K = 128 -> 4 chained) ---
#pragma unroll
    for (int kc = 0; kc < 4; kc++) {
      bf16x16 aV = ldfrag(&PvF[w][kc][0][lane][0], &PvF[w][kc][1][lane][0]);
#pragma unroll
      for (int jo = 0; jo < 4; jo++) {
        bf16x16 bfr = ldfrag(&RVsF[jo][kc][0][lane][0], &RVsF[jo][kc][1][lane][0]);
        accO[jo] = WMMA_BF16(aV, bfr, accO[jo]);
      }
    }
  }

  // out/l, merge heads: [(b*1024+t), h*64+d]
#pragma unroll
  for (int jo = 0; jo < 4; jo++)
#pragma unroll
    for (int g = 0; g < 8; g++) {
      int i = g + 8 * hlf;
      int t = t0 + 16 * w + i;
      int dcol = 16 * jo + n16;
      Og[((size_t)(b * 1024 + t)) * 1024 + h * 64 + dcol] = (__bf16)(accO[jo][g] / l_i[g]);
    }
}

extern "C" void kernel_launch(void* const* d_in, const int* in_sizes, int n_in,
                              void* d_out, int out_size, void* d_ws, size_t ws_size,
                              hipStream_t stream) {
  (void)in_sizes; (void)n_in; (void)out_size; (void)ws_size;
  const float* x  = (const float*)d_in[0];
  const float* Wq = (const float*)d_in[1];
  const float* Wk = (const float*)d_in[2];
  const float* Wv = (const float*)d_in[3];
  const float* Wo = (const float*)d_in[4];
  const float* bo = (const float*)d_in[5];
  const float* rk = (const float*)d_in[6];
  const float* rv = (const float*)d_in[7];
  float* out = (float*)d_out;

  const size_t NX = (size_t)16 * 1024 * 1024;
  const size_t NW = (size_t)1024 * 1024;
  const size_t NRP = (size_t)2080 * 64;

  char* p = (char*)d_ws;
  auto take = [&](size_t elems) {
    __bf16* r = (__bf16*)p;
    p += ((elems * sizeof(__bf16) + 255) / 256) * 256;
    return r;
  };
  __bf16* xbf  = take(NX);
  __bf16* wqT  = take(NW);
  __bf16* wkT  = take(NW);
  __bf16* wvT  = take(NW);
  __bf16* woT  = take(NW);
  __bf16* rkp  = take(NRP);   // [2080][64] zero-padded
  __bf16* rvT  = take(NRP);   // [64][2080] zero-padded
  __bf16* qbf  = take(NX);    // [bh][t][d]
  __bf16* kbf  = take(NX);    // [bh][t][d]
  __bf16* vbfT = take(NX);    // [bh][d][t]
  __bf16* aobf = take(NX);    // [(b*1024+t)][h*64+d]

  cast_f32_to_bf16<<<2048, 256, 0, stream>>>(x, xbf, (int)NX);
  cast_transpose_1024<<<1024, 256, 0, stream>>>(Wq, wqT);
  cast_transpose_1024<<<1024, 256, 0, stream>>>(Wk, wkT);
  cast_transpose_1024<<<1024, 256, 0, stream>>>(Wv, wvT);
  cast_transpose_1024<<<1024, 256, 0, stream>>>(Wo, woT);
  prep_rel<<<520, 256, 0, stream>>>(rk, rv, rkp, rvT);

  dim3 gg(8, 128), gb(256);
  gemm_bf16_128x128<<<gg, gb, 0, stream>>>(xbf, wqT, 16384, 1024, 1024, qbf, nullptr, nullptr, 0.125f, 0);
  gemm_bf16_128x128<<<gg, gb, 0, stream>>>(xbf, wkT, 16384, 1024, 1024, kbf, nullptr, nullptr, 1.0f, 0);
  gemm_bf16_128x128<<<gg, gb, 0, stream>>>(xbf, wvT, 16384, 1024, 1024, vbfT, nullptr, nullptr, 1.0f, 2);

  attn_rel64x32<<<dim3(16, 256), dim3(128), 0, stream>>>(qbf, kbf, vbfT, rkp, rvT, aobf);

  gemm_bf16_128x128<<<gg, gb, 0, stream>>>(aobf, woT, 16384, 1024, 1024, nullptr, out, bo, 1.0f, 1);
}
